// CorrelationLayer_4011499454705
// MI455X (gfx1250) — matrix operand order repaired and checked by reading, not compiled, not run on hardware
//
#include <hip/hip_runtime.h>

// CDNA5 wave32 WMMA operand types
typedef __attribute__((ext_vector_type(16))) __bf16 v16bf;
typedef __attribute__((ext_vector_type(8)))  float  v8f;
typedef __attribute__((ext_vector_type(4)))  unsigned v4u;
typedef __attribute__((ext_vector_type(8)))  unsigned v8u;

#define CC   128      // channels (K)
#define HH   64
#define WW   64
#define BB   4
#define DISP 41       // displacement window side
#define PAD  20
#define NPAD 112      // padded N (7 WMMA tiles)
#define LSTR 105      // ldsC row stride in floats (conflict-free diagonal reads)
#define CH_STRIDE (HH * WW)

// chunk counts for the bf16 pre-converted layouts (1 chunk = 16 halves = 32B)
#define A_CHUNKS (BB * HH * 4 * 2 * WW)     // 131072  -> 4 MB
#define B_CHUNKS (BB * HH * 4 * 2 * NPAD)   // 229376  -> 7 MB
#define B_ROW_BYTES (4 * 2 * NPAD * 16 * 2) // 28672 B per (b,h2) panel

union BfVec16 { __bf16 h[16]; v16bf v; uint4 q[2]; };

// ---------------------------------------------------------------------------
// Phase 1: convert f1 -> A-fragment layout, f2 -> zero-padded B-panel layout.
//   aF1 chunk id = ((((b*HH + h)*4 + kk)*2 + half)*WW + w)
//   bF2 chunk id = ((((b*HH + h2)*4 + kk)*2 + half)*NPAD + wp)
// ---------------------------------------------------------------------------
__global__ __launch_bounds__(256)
void precvt_kernel(const float* __restrict__ f1,
                   const float* __restrict__ f2,
                   __bf16* __restrict__ ws)
{
    const int id = blockIdx.x * 256 + threadIdx.x;
    if (id >= A_CHUNKS + B_CHUNKS) return;

    BfVec16 v;
    if (id < A_CHUNKS) {
        const int w    = id & 63;
        int r          = id >> 6;
        const int half = r & 1;  r >>= 1;
        const int kk   = r & 3;  r >>= 2;
        const int h    = r & 63;
        const int b    = r >> 6;
        const float* p = f1 + ((b * CC) * HH + h) * WW + w;
        const int k0 = 32 * kk + 16 * half;
        const int k1 = 32 * kk + 16 * half + 8;
#pragma unroll
        for (int e = 0; e < 8; ++e) {
            v.h[e]     = (__bf16)p[(k0 + e) * CH_STRIDE];
            v.h[e + 8] = (__bf16)p[(k1 + e) * CH_STRIDE];
        }
    } else {
        const int id2  = id - A_CHUNKS;
        const int wp   = id2 % NPAD;
        int r          = id2 / NPAD;
        const int half = r & 1;  r >>= 1;
        const int kk   = r & 3;  r >>= 2;
        const int h2   = r & 63;
        const int b    = r >> 6;
        const int x    = wp - PAD;
        if (x >= 0 && x < WW) {
            const float* p = f2 + ((b * CC) * HH + h2) * WW + x;
            const int kb = 32 * kk + 16 * half;
#pragma unroll
            for (int e = 0; e < 16; ++e) v.h[e] = (__bf16)p[(kb + e) * CH_STRIDE];
        } else {
#pragma unroll
            for (int e = 0; e < 16; ++e) v.h[e] = (__bf16)0.0f;
        }
    }
    uint4* dst = (uint4*)(ws + (size_t)id * 16);
    dst[0] = v.q[0];
    dst[1] = v.q[1];
}

// ---------------------------------------------------------------------------
// Phase 2: WMMA correlation, A fragments straight from global bf16,
//          B panel DMA'd into LDS by the Tensor Data Mover.
// ---------------------------------------------------------------------------
__global__ __launch_bounds__(128)
void corr_wmma_tdm(const __bf16* __restrict__ ws, float* __restrict__ out)
{
    const int i    = blockIdx.x;
    const int h    = blockIdx.y;
    const int b    = blockIdx.z;
    const int tid  = threadIdx.x;
    const int wave = tid >> 5;
    const int lane = tid & 31;
    const int half = lane >> 4;
    const int l15  = lane & 15;

    const int h2    = h + i - PAD;
    const int obase = ((b * (DISP * DISP) + i * DISP) * HH + h) * WW;

    if (h2 < 0 || h2 >= HH) {
        for (int idx = tid; idx < DISP * WW; idx += 128) {
            const int j = idx >> 6, w = idx & 63;
            __builtin_nontemporal_store(0.0f, &out[obase + j * (HH * WW) + w]);
        }
        return;
    }

    __shared__ __align__(32) __bf16 bB[4 * 2 * NPAD * 16];   // 28 KB
    __shared__ float ldsC[WW * LSTR];                        // 26.9 KB

    const __bf16* aF1 = ws;                                  // A chunks first
    const __bf16* bF2 = ws + (size_t)A_CHUNKS * 16;          // then B panels

    // ---- TDM: copy contiguous 28 KB B panel for (b,h2) into LDS ----
    if (wave == 0) {
        const unsigned long long ga =
            (unsigned long long)(uintptr_t)(bF2) + (unsigned long long)(b * HH + h2) * B_ROW_BYTES;
        const unsigned lds_off = (unsigned)(uintptr_t)&bB[0]; // addr[31:0] = LDS offset
        v4u g0;
        g0[0] = 1u;                                    // count=1, user descriptor
        g0[1] = lds_off;                               // lds_addr
        g0[2] = (unsigned)(ga & 0xFFFFFFFFull);        // global_addr[31:0]
        g0[3] = (unsigned)((ga >> 32) & 0x1FFFFFFull)  // global_addr[56:32]
                | (2u << 30);                          // type = 2 (image)
        v8u g1;
        g1[0] = (3u << 16);                            // data_size=8B, mask=0
        g1[1] = (3584u << 16);                         // tensor_dim0[15:0]=3584
        g1[2] = (1u << 16);                            // dim0[31:16]=0, tensor_dim1=1
        g1[3] = (3584u << 16);                         // dim1[31:16]=0, tile_dim0=3584
        g1[4] = 1u;                                    // tile_dim1=1, tile_dim2=0
        g1[5] = 3584u;                                 // tensor_dim0_stride
        g1[6] = (3584u << 16);                         // tensor_dim1_stride[15:0]
        g1[7] = 0u;
        asm volatile("tensor_load_to_lds %0, %1" :: "s"(g0), "s"(g1) : "memory");
        __builtin_amdgcn_s_wait_tensorcnt(0);
    }

    // ---- A fragments: 2x global_load_b128 per k-step, pre-converted bf16 ----
    const int arow = 16 * wave + l15;
    v16bf afrag[4];
#pragma unroll
    for (int kk = 0; kk < 4; ++kk) {
        const size_t chunk = ((((size_t)(b * HH + h) * 4 + kk) * 2 + half) * WW + arow);
        afrag[kk] = *(const v16bf*)(aF1 + chunk * 16);
    }

    __syncthreads();   // TDM data visible to all waves

    // ---- 7 N-tiles of the padded 64x112 product ----
#pragma unroll
    for (int t = 0; t < 7; ++t) {
        const int wp = 16 * t + l15;
        v8f acc = {};
#pragma unroll
        for (int kk = 0; kk < 4; ++kk) {
            const v16bf bfrag =
                *(const v16bf*)&bB[((kk * 2 + half) * NPAD + wp) * 16];
            acc = __builtin_amdgcn_wmma_f32_16x16x32_bf16(
                    false, afrag[kk], false, bfrag,
                    (short)0, acc, false, false);
        }
#pragma unroll
        for (int p = 0; p < 8; ++p) {
            const int w = 16 * wave + p + 8 * half;
            ldsC[w * LSTR + wp] = acc[p];
        }
    }

    __syncthreads();

    const int w = tid & 63;
    for (int j = tid >> 6; j < DISP; j += 2) {
        __builtin_nontemporal_store(ldsC[w * LSTR + w + j],
                                    &out[obase + j * (HH * WW) + w]);
    }
}

// ---------------------------------------------------------------------------
// Fallback (ws too small): fused convert-in-kernel version (round-2 design).
// ---------------------------------------------------------------------------
typedef __attribute__((ext_vector_type(8))) __bf16 v8bf;
union BfVec8 { __bf16 h[8]; v8bf v; uint4 q; };

__global__ __launch_bounds__(128)
void corr_wmma_fused(const float* __restrict__ f1,
                     const float* __restrict__ f2,
                     float* __restrict__ out)
{
    const int i    = blockIdx.x;
    const int h    = blockIdx.y;
    const int b    = blockIdx.z;
    const int tid  = threadIdx.x;
    const int wave = tid >> 5;
    const int lane = tid & 31;
    const int half = lane >> 4;
    const int l15  = lane & 15;

    const int h2    = h + i - PAD;
    const int obase = ((b * (DISP * DISP) + i * DISP) * HH + h) * WW;

    if (h2 < 0 || h2 >= HH) {
        for (int idx = tid; idx < DISP * WW; idx += 128) {
            const int j = idx >> 6, w = idx & 63;
            __builtin_nontemporal_store(0.0f, &out[obase + j * (HH * WW) + w]);
        }
        return;
    }

    __shared__ __align__(32) __bf16 bB[4 * 2 * NPAD * 16];
    __shared__ float ldsC[WW * LSTR];

    const int    arow   = 16 * wave + l15;
    const float* a_base = f1 + ((b * CC) * HH + h) * WW + arow;
    v16bf afrag[4];
#pragma unroll
    for (int kk = 0; kk < 4; ++kk) {
        BfVec16 av;
        const int k0 = 32 * kk + 16 * half;
        const int k1 = 32 * kk + 16 * half + 8;
#pragma unroll
        for (int e = 0; e < 8; ++e) {
            av.h[e]     = (__bf16)a_base[(k0 + e) * CH_STRIDE];
            av.h[e + 8] = (__bf16)a_base[(k1 + e) * CH_STRIDE];
        }
        afrag[kk] = av.v;
    }

    const float* b_row = f2 + ((b * CC) * HH + h2) * WW;
    for (int z = tid; z < 8 * 48 * 2; z += 128) {
        const int s   = z >> 1;
        const int run = s / 48;
        const int ci  = s % 48;
        const int wp  = (ci < 20) ? ci : (ci + 64);
        *(uint4*)&bB[(run * NPAD + wp) * 16 + (z & 1) * 8] = uint4{0, 0, 0, 0};
    }
    for (int o = tid; o < (CC / 8) * WW; o += 128) {
        const int x  = o & 63;
        const int c0 = (o >> 6) << 3;
        const float* p = b_row + c0 * CH_STRIDE + x;
        BfVec8 vv;
#pragma unroll
        for (int e = 0; e < 8; ++e) vv.h[e] = (__bf16)p[e * CH_STRIDE];
        const int kk = c0 >> 5;
        const int hf = (c0 >> 4) & 1;
        const int e0 = c0 & 15;
        const int wp = x + PAD;
        *(uint4*)&bB[((kk * 2 + hf) * NPAD + wp) * 16 + e0] = vv.q;
    }

    __syncthreads();

#pragma unroll
    for (int t = 0; t < 7; ++t) {
        const int wp = 16 * t + l15;
        v8f acc = {};
#pragma unroll
        for (int kk = 0; kk < 4; ++kk) {
            const v16bf bfrag =
                *(const v16bf*)&bB[((kk * 2 + half) * NPAD + wp) * 16];
            acc = __builtin_amdgcn_wmma_f32_16x16x32_bf16(
                    false, afrag[kk], false, bfrag,
                    (short)0, acc, false, false);
        }
#pragma unroll
        for (int p = 0; p < 8; ++p) {
            const int w = 16 * wave + p + 8 * half;
            ldsC[w * LSTR + wp] = acc[p];
        }
    }

    __syncthreads();

    const int w = tid & 63;
    for (int j = tid >> 6; j < DISP; j += 2) {
        __builtin_nontemporal_store(ldsC[w * LSTR + w + j],
                                    &out[obase + j * (HH * WW) + w]);
    }
}

extern "C" void kernel_launch(void* const* d_in, const int* in_sizes, int n_in,
                              void* d_out, int out_size, void* d_ws, size_t ws_size,
                              hipStream_t stream) {
    const float* f1 = (const float*)d_in[0];
    const float* f2 = (const float*)d_in[1];
    float* out = (float*)d_out;
    (void)in_sizes; (void)n_in; (void)out_size;

    const size_t ws_needed = (size_t)(A_CHUNKS + B_CHUNKS) * 32;
    dim3 grid(DISP, HH, BB);

    if (ws_size >= ws_needed) {
        __bf16* ws = (__bf16*)d_ws;
        const int nchunk = A_CHUNKS + B_CHUNKS;
        precvt_kernel<<<(nchunk + 255) / 256, 256, 0, stream>>>(f1, f2, ws);
        corr_wmma_tdm<<<grid, 128, 0, stream>>>(ws, out);
    } else {
        corr_wmma_fused<<<grid, 128, 0, stream>>>(f1, f2, out);
    }
}
